// MLPPredictor_69965017252748
// MI455X (gfx1250) — matrix operand order repaired and checked by reading, not compiled, not run on hardware
//
#include <hip/hip_runtime.h>
#include <hip/hip_bf16.h>

typedef __attribute__((ext_vector_type(16))) __bf16 v16bf;
typedef __attribute__((ext_vector_type(8)))  __bf16 v8bf;
typedef __attribute__((ext_vector_type(8)))  float  v8f;

#define HFEAT   128
#define TWO_H   256
#define NTYPES  8
#define NEDGES  500000
#define NTILES  (NEDGES / 16)   // 31250, exact

// LDS element layout (bf16 elements):
//   [0, 32768)            W1 B-fragments: frag f = nt*8+kt, [f][lane][16]
//   [32768, 34816)        W2 B-fragments: frag kt,          [kt][lane][16]
//   [34816, 34816+wid*2048) per-wave hidden staging, 16x128 row-major
#define W1_ELEMS (64 * 32 * 16)
#define W2_ELEMS (4 * 32 * 16)
#define HID_ELEMS (16 * 128)

__device__ inline void cvt8(v16bf& a, int base, float4 x, float4 y) {
    a[base + 0] = (__bf16)x.x; a[base + 1] = (__bf16)x.y;
    a[base + 2] = (__bf16)x.z; a[base + 3] = (__bf16)x.w;
    a[base + 4] = (__bf16)y.x; a[base + 5] = (__bf16)y.y;
    a[base + 6] = (__bf16)y.z; a[base + 7] = (__bf16)y.w;
}

__global__ __launch_bounds__(256) void edge_mlp_wmma_kernel(
    const float* __restrict__ h,
    const int*   __restrict__ src,
    const int*   __restrict__ dst,
    const float* __restrict__ W1,
    const float* __restrict__ b1,
    const float* __restrict__ W2,
    const float* __restrict__ b2,
    float* __restrict__ out)
{
    extern __shared__ __align__(32) __bf16 smem[];
    __bf16* lds_w1  = smem;
    __bf16* lds_w2  = smem + W1_ELEMS;
    __bf16* lds_hid = smem + W1_ELEMS + W2_ELEMS;

    const int tid = threadIdx.x;

    // ---- Pack W1 into WMMA-B fragment layout (bf16), once per block ----
    // idx -> i = idx&15 (element), l = (idx>>4)&31 (lane), f = idx>>9 (fragment)
    // B[k][n] = W1[n][k];  n = nt*16 + (l&15);  k = kt*32 + (l>>4)*8 + (i>>3)*16 + (i&7)
    for (int idx = tid; idx < W1_ELEMS; idx += blockDim.x) {
        int i  = idx & 15;
        int l  = (idx >> 4) & 31;
        int f  = idx >> 9;
        int nt = f >> 3, kt = f & 7;
        int n  = nt * 16 + (l & 15);
        int k  = kt * 32 + ((l >> 4) << 3) + ((i >> 3) << 4) + (i & 7);
        lds_w1[idx] = (__bf16)W1[n * TWO_H + k];
    }
    // ---- Pack W2 (N padded 8 -> 16 with zeros) ----
    for (int idx = tid; idx < W2_ELEMS; idx += blockDim.x) {
        int i  = idx & 15;
        int l  = (idx >> 4) & 31;
        int kt = idx >> 9;
        int n  = l & 15;
        int k  = kt * 32 + ((l >> 4) << 3) + ((i >> 3) << 4) + (i & 7);
        lds_w2[idx] = (n < NTYPES) ? (__bf16)W2[n * HFEAT + k] : (__bf16)0.0f;
    }
    __syncthreads();

    const int lane       = tid & 31;
    const int wid        = tid >> 5;
    const int nwaves_blk = blockDim.x >> 5;
    const int gwave      = blockIdx.x * nwaves_blk + wid;
    const int nwaves     = gridDim.x * nwaves_blk;

    __bf16* hid = lds_hid + wid * HID_ELEMS;

    const int m    = lane & 15;   // row within 16-edge tile (A layout) / N column (D layout)
    const int half = lane >> 4;

    for (int tile = gwave; tile < NTILES; tile += nwaves) {
        const int e0   = tile * 16;
        const int edge = e0 + m;
        const float* rowS = h + (long)src[edge] * HFEAT;
        const float* rowD = h + (long)dst[edge] * HFEAT;

        // ---- Build 8 A-fragments (16 edges x 256 concat features, bf16) ----
        // ISA 16-bit A layout: half 0 -> K kb+0..7 (elems 0-7), kb+16..23 (elems 8-15)
        //                      half 1 -> K kb+8..15,            kb+24..31
        v16bf a[8];
        #pragma unroll
        for (int kt = 0; kt < 8; ++kt) {
            const float* row = (kt < 4) ? rowS : rowD;
            int kb  = (kt & 3) * 32;
            int flo = kb + half * 8;
            int fhi = kb + 16 + half * 8;
            float4 lo0 = *(const float4*)(row + flo);
            float4 lo1 = *(const float4*)(row + flo + 4);
            float4 hi0 = *(const float4*)(row + fhi);
            float4 hi1 = *(const float4*)(row + fhi + 4);
            cvt8(a[kt], 0, lo0, lo1);
            cvt8(a[kt], 8, hi0, hi1);
        }

        // ---- Layer 1: hidden[16x128] = relu(e @ W1^T + b1), stage to LDS ----
        #pragma unroll
        for (int nt = 0; nt < 8; ++nt) {
            v8f c = {};
            #pragma unroll
            for (int kt = 0; kt < 8; ++kt) {
                v16bf b = *(const v16bf*)(lds_w1 + ((nt * 8 + kt) * 32 + lane) * 16);
                c = __builtin_amdgcn_wmma_f32_16x16x32_bf16(
                        false, a[kt], false, b, (short)0, c, false, false);
            }
            float bias = b1[nt * 16 + m];
            // D layout: VGPR r holds M=r (lanes 0-15) / M=r+8 (lanes 16-31), N = nt*16+m
            #pragma unroll
            for (int r = 0; r < 8; ++r) {
                float v = c[r] + bias;
                v = v > 0.0f ? v : 0.0f;
                hid[(r + 8 * half) * HFEAT + nt * 16 + m] = (__bf16)v;
            }
        }

        // ---- Layer 2: logits[16x16pad] = hidden @ W2pad^T ----
        v8f c2 = {};
        #pragma unroll
        for (int kt = 0; kt < 4; ++kt) {
            int kb = kt * 32;
            v8bf lo = *(const v8bf*)(hid + m * HFEAT + kb + half * 8);
            v8bf hi = *(const v8bf*)(hid + m * HFEAT + kb + 16 + half * 8);
            v16bf a2;
            #pragma unroll
            for (int i = 0; i < 8; ++i) { a2[i] = lo[i]; a2[8 + i] = hi[i]; }
            v16bf b = *(const v16bf*)(lds_w2 + (kt * 32 + lane) * 16);
            c2 = __builtin_amdgcn_wmma_f32_16x16x32_bf16(
                     false, a2, false, b, (short)0, c2, false, false);
        }

        // ---- Bias + softmax over 8 types (cross-lane within groups 0-7 / 16-23) ----
        float bias2 = (m < NTYPES) ? b2[m] : 0.0f;
        float vals[8];
        #pragma unroll
        for (int r = 0; r < 8; ++r) {
            float v = c2[r] + bias2;
            float mx = v;
            mx = fmaxf(mx, __shfl_xor(mx, 1, 32));
            mx = fmaxf(mx, __shfl_xor(mx, 2, 32));
            mx = fmaxf(mx, __shfl_xor(mx, 4, 32));
            float e = __expf(v - mx);
            float s = e;
            s += __shfl_xor(s, 1, 32);
            s += __shfl_xor(s, 2, 32);
            s += __shfl_xor(s, 4, 32);
            vals[r] = e / s;
        }

        // ---- Store: out[(e0 + M) * 8 + type], type = m (lanes with m < 8) ----
        if (m < NTYPES) {
            #pragma unroll
            for (int r = 0; r < 8; ++r) {
                out[(long)(e0 + r + 8 * half) * NTYPES + m] = vals[r];
            }
        }
    }
}

extern "C" void kernel_launch(void* const* d_in, const int* in_sizes, int n_in,
                              void* d_out, int out_size, void* d_ws, size_t ws_size,
                              hipStream_t stream) {
    const float* h   = (const float*)d_in[0];
    const int*   src = (const int*)d_in[1];
    const int*   dst = (const int*)d_in[2];
    const float* W1  = (const float*)d_in[3];
    const float* b1  = (const float*)d_in[4];
    const float* W2  = (const float*)d_in[5];
    const float* b2  = (const float*)d_in[6];
    float* out = (float*)d_out;

    dim3 grid(1024), block(256);
    size_t shmem = (size_t)(W1_ELEMS + W2_ELEMS + 8 * HID_ELEMS) * sizeof(__bf16); // 100 KB
    edge_mlp_wmma_kernel<<<grid, block, shmem, stream>>>(h, src, dst, W1, b1, W2, b2, out);
}
